// TransformerEncoderLayer_21509196219147
// MI455X (gfx1250) — matrix operand order repaired and checked by reading, not compiled, not run on hardware
//
#include <hip/hip_runtime.h>
#include <hip/hip_bf16.h>
#include <math.h>

typedef __attribute__((ext_vector_type(16))) _Float16 v16h;
typedef __attribute__((ext_vector_type(8)))  float    v8f;
typedef int v4i_ __attribute__((vector_size(16)));   // matches builtin param type
typedef _Float16 f16;

#define DM    512
#define DFF   2048
#define NB    8
#define NC    64
#define NP    128
#define NTOK  (NB*NC*NP)     /* 65536 tokens */
#define HEADS 4
#define HD    64
#define RHALF 32             /* rope half of head dim */

#define BM 128
#define BN 128
#define BK 64

#if defined(__gfx1250__) && __has_builtin(__builtin_amdgcn_global_load_async_to_lds_b128)
#define USE_ASYNC_LDS 1
#else
#define USE_ASYNC_LDS 0
#endif

#define AS1 __attribute__((address_space(1)))
#define AS3 __attribute__((address_space(3)))

__device__ __forceinline__ void wait_asynccnt0() {
#if __has_builtin(__builtin_amdgcn_s_wait_asynccnt)
  __builtin_amdgcn_s_wait_asynccnt(0);
#else
  asm volatile("s_wait_asynccnt 0x0" ::: "memory");
#endif
}

// ---------------------------------------------------------------------------
// f32 -> f16 convert
// ---------------------------------------------------------------------------
__global__ void k_f32_to_f16(const float* __restrict__ in, f16* __restrict__ out, int n) {
  int i = blockIdx.x * blockDim.x + threadIdx.x;
  if (i < n) out[i] = (f16)in[i];
}

// ---------------------------------------------------------------------------
// LayerNorm over D=512, writes f16 (feeds WMMA GEMMs)
// ---------------------------------------------------------------------------
__global__ __launch_bounds__(128)
void k_layernorm(const float* __restrict__ x, const float* __restrict__ w,
                 const float* __restrict__ b, f16* __restrict__ out) {
  __shared__ float s1[128], s2[128];
  size_t t = blockIdx.x;
  const float* xr = x + t * DM;
  float v[4], sum = 0.f, sq = 0.f;
  #pragma unroll
  for (int i = 0; i < 4; i++) {
    v[i] = xr[threadIdx.x * 4 + i];
    sum += v[i]; sq += v[i] * v[i];
  }
  s1[threadIdx.x] = sum; s2[threadIdx.x] = sq;
  __syncthreads();
  for (int off = 64; off > 0; off >>= 1) {
    if (threadIdx.x < off) {
      s1[threadIdx.x] += s1[threadIdx.x + off];
      s2[threadIdx.x] += s2[threadIdx.x + off];
    }
    __syncthreads();
  }
  float mean = s1[0] / DM;
  float var  = s2[0] / DM - mean * mean;
  float rstd = rsqrtf(var + 1e-5f);
  #pragma unroll
  for (int i = 0; i < 4; i++) {
    int c = threadIdx.x * 4 + i;
    out[t * DM + c] = (f16)((v[i] - mean) * rstd * w[c] + b[c]);
  }
}

// ---------------------------------------------------------------------------
// Coord-RoPE table: cos/sin [C=64][32] from coords (center, normalize by max r)
// ---------------------------------------------------------------------------
__global__ __launch_bounds__(64)
void k_rope_spatial(const float* __restrict__ coords,
                    float* __restrict__ cosS, float* __restrict__ sinS) {
  __shared__ float sx[64], sy[64], sz[64], rr[64];
  int c = threadIdx.x;
  sx[c] = coords[c * 3 + 0];
  sy[c] = coords[c * 3 + 1];
  sz[c] = coords[c * 3 + 2];
  __syncthreads();
  float mx = 0.f, my = 0.f, mz = 0.f;
  for (int i = 0; i < 64; i++) { mx += sx[i]; my += sy[i]; mz += sz[i]; }
  mx /= 64.f; my /= 64.f; mz /= 64.f;
  float cx = sx[c] - mx, cy = sy[c] - my, cz = sz[c] - mz;
  rr[c] = sqrtf(cx * cx + cy * cy + cz * cz);
  __syncthreads();
  float rmax = 0.f;
  for (int i = 0; i < 64; i++) rmax = fmaxf(rmax, rr[i]);
  float inv = 1.0f / fmaxf(rmax, 1e-6f);
  cx *= inv; cy *= inv; cz *= inv;
  const float TWO_PI = 6.28318530717958647692f;
  for (int j = 0; j < RHALF; j++) {
    int axis = j % 3;
    float val = (axis == 0) ? cx : (axis == 1) ? cy : cz;
    float freq = __powf(10000.0f, -(float)j / (float)RHALF);
    float th = TWO_PI * val * freq;
    cosS[c * RHALF + j] = __cosf(th);
    sinS[c * RHALF + j] = __sinf(th);
  }
}

// ---------------------------------------------------------------------------
// WMMA GEMM:  C[M x N] = A[M x K] * W[N x K]^T + bias, epilogue variants.
//   OP 0: f32 out    OP 1: f16 out    OP 2: gelu -> f16 out
//   OP 3: f32 out + residual add
// Block tile 128(M) x 128(N), K-stage 64, 8 waves (wave32), wave tile 32x64:
// 2x4 fragments -> 16 v_wmma_f32_16x16x32_f16 per stage.  A/B staged into
// 32KB LDS via GLOBAL_LOAD_ASYNC_TO_LDS_B128 (ASYNCcnt; no VGPR bounce, no
// per-chunk waitcnt chain); next K-stage prefetched (global_prefetch_b8).
// Fragment gathers are contiguous 16B ds reads matching the ISA A/B VGPR
// layouts (A: lanes16-31 hold K+8; B: same).
// Requires M%128==0, N%128==0, K%64==0 (true for all shapes here).
// ---------------------------------------------------------------------------
template<int OP>
__global__ __launch_bounds__(256)
void k_wmma_gemm(const f16* __restrict__ A, int lda,
                 const f16* __restrict__ W, int K,
                 const float* __restrict__ bias,
                 float* __restrict__ Cf, f16* __restrict__ Ch, int ldc,
                 const float* __restrict__ Res, int ldr) {
  __shared__ f16 sA[BM * BK];   // 16 KB
  __shared__ f16 sB[BN * BK];   // 16 KB
  const int tid  = threadIdx.x;
  const int lane = tid & 31;
  const int wid  = tid >> 5;
  const int wm   = wid & 3;          // 4 waves along M: 32 rows each
  const int wn   = wid >> 2;         // 2 waves along N: 64 cols each
  const size_t rowBase = (size_t)blockIdx.x * BM;
  const int    nBase   = blockIdx.y * BN;

  v8f acc[2][4] = {};

  const int hi   = (lane >> 4) & 1;  // lane group (0: lanes 0-15, 1: 16-31)
  const int koff = hi * 8;           // A frag: upper lanes hold K+8
  const int kblk = hi * 16;          // unused; B frag also uses koff (K+8)

  const int sr   = tid >> 1;         // staging row (0..127)
  const int sp   = (tid & 1) * 32;   // staging column chunk (halves)

  for (int k0 = 0; k0 < K; k0 += BK) {
    const f16* gA = A + (rowBase + sr) * (size_t)lda + k0 + sp;
    const f16* gW = W + (size_t)(nBase + sr) * K + k0 + sp;
    f16* lA = &sA[sr * BK + sp];
    f16* lB = &sB[sr * BK + sp];
#if USE_ASYNC_LDS
    // ---- async DMA global -> LDS, 16B per op, 8 ops in flight, one wait
    #pragma unroll
    for (int u = 0; u < 4; u++) {
      __builtin_amdgcn_global_load_async_to_lds_b128(
          (AS1 v4i_*)(uintptr_t)(gA + u * 8),
          (AS3 v4i_*)(uint32_t)(uintptr_t)(lA + u * 8), 0, 0);
      __builtin_amdgcn_global_load_async_to_lds_b128(
          (AS1 v4i_*)(uintptr_t)(gW + u * 8),
          (AS3 v4i_*)(uint32_t)(uintptr_t)(lB + u * 8), 0, 0);
    }
#else
    {
      const uint4* s = reinterpret_cast<const uint4*>(gA);
      uint4* d = reinterpret_cast<uint4*>(lA);
      d[0] = s[0]; d[1] = s[1]; d[2] = s[2]; d[3] = s[3];
    }
    {
      const uint4* s = reinterpret_cast<const uint4*>(gW);
      uint4* d = reinterpret_cast<uint4*>(lB);
      d[0] = s[0]; d[1] = s[1]; d[2] = s[2]; d[3] = s[3];
    }
#endif
    // ---- prefetch next K stage into L2/WGP$ while this one computes
    if (k0 + BK < K) {
      __builtin_prefetch(gA + BK, 0, 1);
      __builtin_prefetch(gW + BK, 0, 1);
    }
#if USE_ASYNC_LDS
    wait_asynccnt0();
#endif
    __syncthreads();

    #pragma unroll
    for (int kk = 0; kk < 2; kk++) {            // two 32-K chunks per stage
      union U { uint4 q[2]; v16h v; };
      U fa[2], fb[4];
      #pragma unroll
      for (int fm = 0; fm < 2; fm++) {
        int row = wm * 32 + fm * 16 + (lane & 15);
        fa[fm].q[0] = *reinterpret_cast<const uint4*>(&sA[row * BK + kk * 32 + koff]);
        fa[fm].q[1] = *reinterpret_cast<const uint4*>(&sA[row * BK + kk * 32 + 16 + koff]);
      }
      #pragma unroll
      for (int fn = 0; fn < 4; fn++) {
        int n = wn * 64 + fn * 16 + (lane & 15);
        fb[fn].q[0] = *reinterpret_cast<const uint4*>(&sB[n * BK + kk * 32 + koff]);
        fb[fn].q[1] = *reinterpret_cast<const uint4*>(&sB[n * BK + kk * 32 + 16 + koff]);
      }
      #pragma unroll
      for (int fm = 0; fm < 2; fm++)
        #pragma unroll
        for (int fn = 0; fn < 4; fn++)
          acc[fm][fn] = __builtin_amdgcn_wmma_f32_16x16x32_f16(
              false, fa[fm].v, false, fb[fn].v, (short)0, acc[fm][fn], false, false);
    }
    __syncthreads();
  }

  // ---- epilogue: lane = N + 16*(M>=8); vgpr r -> row (M&7)
  #pragma unroll
  for (int fm = 0; fm < 2; fm++) {
    #pragma unroll
    for (int fn = 0; fn < 4; fn++) {
      int    nl   = nBase + wn * 64 + fn * 16 + (lane & 15);
      float  bv   = bias[nl];
      size_t mrow = rowBase + (size_t)(wm * 32 + fm * 16 + hi * 8);
      #pragma unroll
      for (int r = 0; r < 8; r++) {
        float v = acc[fm][fn][r] + bv;
        size_t idx = (mrow + r) * (size_t)ldc + nl;
        if (OP == 0) {
          Cf[idx] = v;
        } else if (OP == 1) {
          Ch[idx] = (f16)v;
        } else if (OP == 2) {
          Ch[idx] = (f16)(0.5f * v * (1.0f + erff(v * 0.70710678118654752440f)));
        } else {
          Cf[idx] = v + Res[(mrow + r) * (size_t)ldr + nl];
        }
      }
    }
  }
}

// ---------------------------------------------------------------------------
// MHA (small seq): one block per (sequence, head), one query row per lane.
// K/V staged (RoPE'd) in LDS; online softmax, scale = 1/sqrt(64).
// MODE 0: spatial  (seq over C, stride P tokens, cos/sin table)   L = 64
// MODE 1: temporal (seq over P, stride 1, index rope inline)      L = 128
// qkv: [NTOK][768] f16 (q|k|v each 256 = 4 heads x 64).
// out: [NTOK][512] f16, head slice written at colOff + h*64.
// ---------------------------------------------------------------------------
template<int L, int MODE>
__global__ __launch_bounds__(L)
void k_attention(const f16* __restrict__ qkv,
                 const float* __restrict__ cosT, const float* __restrict__ sinT,
                 f16* __restrict__ out, int colOff) {
  __shared__ f16 ksh[L][HD];
  __shared__ f16 vsh[L][HD];
  const int j  = threadIdx.x;     // sequence position owned by this lane
  const int h  = blockIdx.y;
  const int bb = blockIdx.x;
  size_t base, stride;
  if (MODE == 0) { size_t b = bb / NP, p = bb % NP; base = b * NC * NP + p; stride = NP; }
  else           { base = (size_t)bb * NP;                                 stride = 1;  }

  float cs[RHALF], sn[RHALF];
  if (MODE == 0) {
    #pragma unroll
    for (int i = 0; i < RHALF; i++) { cs[i] = cosT[j * RHALF + i]; sn[i] = sinT[j * RHALF + i]; }
  } else {
    #pragma unroll
    for (int i = 0; i < RHALF; i++) {
      float freq = __powf(10000.0f, -(float)i / (float)RHALF);
      float th = (float)j * freq;
      cs[i] = __cosf(th); sn[i] = __sinf(th);
    }
  }

  const size_t t = base + (size_t)j * stride;
  const f16* qp = qkv + t * 768 +       h * HD;
  const f16* kp = qkv + t * 768 + 256 + h * HD;
  const f16* vp = qkv + t * 768 + 512 + h * HD;

  float q[HD];
  #pragma unroll
  for (int i = 0; i < RHALF; i++) {
    float q1 = (float)qp[i], q2 = (float)qp[i + RHALF];
    float k1 = (float)kp[i], k2 = (float)kp[i + RHALF];
    q[i]         = q1 * cs[i] - q2 * sn[i];
    q[i + RHALF] = q1 * sn[i] + q2 * cs[i];
    ksh[j][i]         = (f16)(k1 * cs[i] - k2 * sn[i]);
    ksh[j][i + RHALF] = (f16)(k1 * sn[i] + k2 * cs[i]);
  }
  #pragma unroll
  for (int i = 0; i < HD; i++) vsh[j][i] = vp[i];
  __syncthreads();

  float m = -1e30f, l = 0.0f, acc[HD];
  #pragma unroll
  for (int i = 0; i < HD; i++) acc[i] = 0.0f;

  for (int jj = 0; jj < L; jj++) {
    float s = 0.0f;
    #pragma unroll
    for (int d = 0; d < HD; d++) s += q[d] * (float)ksh[jj][d];
    s *= 0.125f;                       // 64^-0.5
    float nm = fmaxf(m, s);
    float f  = __expf(m - nm);
    float e  = __expf(s - nm);
    l = l * f + e;
    #pragma unroll
    for (int d = 0; d < HD; d++) acc[d] = acc[d] * f + e * (float)vsh[jj][d];
    m = nm;
  }
  float inv = 1.0f / l;
  f16* op = out + t * 512 + colOff + h * HD;
  #pragma unroll
  for (int d = 0; d < HD; d++) op[d] = (f16)(acc[d] * inv);
}

// ---------------------------------------------------------------------------
// Host orchestration
// ---------------------------------------------------------------------------
extern "C" void kernel_launch(void* const* d_in, const int* in_sizes, int n_in,
                              void* d_out, int out_size, void* d_ws, size_t ws_size,
                              hipStream_t stream) {
  const float* src         = (const float*)d_in[0];
  const float* coords      = (const float*)d_in[1];
  // d_in[2] q_attn, d_in[3] rms_attn: dead (softmax over singleton axis == 1)
  const float* proj_attn_w = (const float*)d_in[4];
  const float* proj_attn_b = (const float*)d_in[5];
  // d_in[6] q_ffn, d_in[7] rms_ffn: dead
  const float* proj_ffn_w  = (const float*)d_in[8];
  const float* proj_ffn_b  = (const float*)d_in[9];
  const float* ln1_w = (const float*)d_in[10];
  const float* ln1_b = (const float*)d_in[11];
  const float* ln2_w = (const float*)d_in[12];
  const float* ln2_b = (const float*)d_in[13];
  const float* qkv_s_w = (const float*)d_in[14];
  const float* qkv_s_b = (const float*)d_in[15];
  const float* out_s_w = (const float*)d_in[16];
  const float* out_s_b = (const float*)d_in[17];
  const float* qkv_t_w = (const float*)d_in[18];
  const float* qkv_t_b = (const float*)d_in[19];
  const float* out_t_w = (const float*)d_in[20];
  const float* out_t_b = (const float*)d_in[21];
  const float* w1 = (const float*)d_in[22];
  const float* b1 = (const float*)d_in[23];
  const float* w2 = (const float*)d_in[24];
  const float* b2 = (const float*)d_in[25];

  char* wsp = (char*)d_ws;
  auto alloc = [&](size_t bytes) -> char* {
    char* p = wsp; wsp += (bytes + 255) & ~(size_t)255; return p;
  };
  f16*  h_pa  = (f16*)alloc((size_t)DM * DM * 2);
  f16*  h_pf  = (f16*)alloc((size_t)DM * DM * 2);
  f16*  h_qs  = (f16*)alloc((size_t)768 * 256 * 2);
  f16*  h_qt  = (f16*)alloc((size_t)768 * 256 * 2);
  f16*  h_os  = (f16*)alloc((size_t)256 * 256 * 2);
  f16*  h_ot  = (f16*)alloc((size_t)256 * 256 * 2);
  f16*  h_w1  = (f16*)alloc((size_t)DFF * DM * 2);
  f16*  h_w2  = (f16*)alloc((size_t)DM * DFF * 2);
  float* cosS = (float*)alloc((size_t)NC * RHALF * 4);
  float* sinS = (float*)alloc((size_t)NC * RHALF * 4);
  f16*  A16   = (f16*)alloc((size_t)NTOK * DM * 2);    // f16 GEMM input (reused)
  f16*  QKV   = (f16*)alloc((size_t)NTOK * 768 * 2);   // qkv (spatial then temporal)
  f16*  AO    = (f16*)alloc((size_t)NTOK * DM * 2);    // attn out (ys | yt)
  float* res  = (float*)alloc((size_t)NTOK * DM * 4);  // res1 then res2
  float* xf   = (float*)alloc((size_t)NTOK * DM * 4);  // x after attention block
  f16*  H16   = (f16*)alloc((size_t)NTOK * DFF * 2);   // gelu(ffn1) hidden

  auto cvt = [&](const float* a, f16* b, size_t n) {
    k_f32_to_f16<<<dim3((unsigned)((n + 255) / 256)), 256, 0, stream>>>(a, b, (int)n);
  };
  auto gemm = [&](int OP, const f16* A, int lda, const f16* W, int K,
                  const float* bias, int M, int N,
                  float* Cf, f16* Ch, int ldc, const float* Rp, int ldr) {
    dim3 g(M / BM, N / BN);
    switch (OP) {
      case 0: k_wmma_gemm<0><<<g, 256, 0, stream>>>(A, lda, W, K, bias, Cf, Ch, ldc, Rp, ldr); break;
      case 1: k_wmma_gemm<1><<<g, 256, 0, stream>>>(A, lda, W, K, bias, Cf, Ch, ldc, Rp, ldr); break;
      case 2: k_wmma_gemm<2><<<g, 256, 0, stream>>>(A, lda, W, K, bias, Cf, Ch, ldc, Rp, ldr); break;
      default: k_wmma_gemm<3><<<g, 256, 0, stream>>>(A, lda, W, K, bias, Cf, Ch, ldc, Rp, ldr); break;
    }
  };

  // ---- weight conversions + rope table
  cvt(proj_attn_w, h_pa, (size_t)DM * DM);
  cvt(proj_ffn_w,  h_pf, (size_t)DM * DM);
  cvt(qkv_s_w,     h_qs, (size_t)768 * 256);
  cvt(qkv_t_w,     h_qt, (size_t)768 * 256);
  cvt(out_s_w,     h_os, (size_t)256 * 256);
  cvt(out_t_w,     h_ot, (size_t)256 * 256);
  cvt(w1,          h_w1, (size_t)DFF * DM);
  cvt(w2,          h_w2, (size_t)DM * DFF);
  k_rope_spatial<<<1, 64, 0, stream>>>(coords, cosS, sinS);

  // ---- res1 = src @ proj_attn_w^T + b
  cvt(src, A16, (size_t)NTOK * DM);
  gemm(0, A16, DM, h_pa, DM, proj_attn_b, NTOK, DM, res, nullptr, DM, nullptr, 0);

  // ---- x_ln = LN1(src) (f16, overwrites src16 after res1 GEMM drained)
  k_layernorm<<<NTOK, 128, 0, stream>>>(src, ln1_w, ln1_b, A16);

  // ---- spatial branch: qkv on first half, attn over C
  gemm(1, A16, DM, h_qs, 256, qkv_s_b, NTOK, 768, nullptr, QKV, 768, nullptr, 0);
  k_attention<64, 0><<<dim3(NB * NP, HEADS), 64, 0, stream>>>(QKV, cosS, sinS, AO, 0);

  // ---- temporal branch: qkv on second half, attn over P (QKV reused)
  gemm(1, A16 + 256, DM, h_qt, 256, qkv_t_b, NTOK, 768, nullptr, QKV, 768, nullptr, 0);
  k_attention<128, 1><<<dim3(NB * NC, HEADS), 128, 0, stream>>>(QKV, nullptr, nullptr, AO, 256);

  // ---- x = res1 + [ys@out_s^T+b | yt@out_t^T+b]  (residual fused in epilogue)
  gemm(3, AO,       DM, h_os, 256, out_s_b, NTOK, 256, xf,       nullptr, DM, res,       DM);
  gemm(3, AO + 256, DM, h_ot, 256, out_t_b, NTOK, 256, xf + 256, nullptr, DM, res + 256, DM);

  // ---- res2 = x @ proj_ffn_w^T + b   (reuse res buffer)
  cvt(xf, A16, (size_t)NTOK * DM);
  gemm(0, A16, DM, h_pf, DM, proj_ffn_b, NTOK, DM, res, nullptr, DM, nullptr, 0);

  // ---- FFN: hidden = gelu(LN2(x) @ w1^T + b1); out = res2 + hidden @ w2^T + b2
  k_layernorm<<<NTOK, 128, 0, stream>>>(xf, ln2_w, ln2_b, A16);
  gemm(2, A16, DM, h_w1, DM, b1, NTOK, DFF, nullptr, H16, DFF, nullptr, 0);
  gemm(3, H16, DFF, h_w2, DFF, b2, NTOK, DM, (float*)d_out, nullptr, DM, res, DM);

  (void)in_sizes; (void)n_in; (void)out_size; (void)ws_size;
}
